// MultiHeadAttention_25220047962714
// MI455X (gfx1250) — matrix operand order repaired and checked by reading, not compiled, not run on hardware
//
#include <hip/hip_runtime.h>

typedef __attribute__((ext_vector_type(16))) _Float16 v16h;
typedef __attribute__((ext_vector_type(8)))  _Float16 v8h;
typedef __attribute__((ext_vector_type(8)))  float    v8f;
typedef __attribute__((ext_vector_type(4)))  unsigned int v4u;
typedef __attribute__((ext_vector_type(8)))  int v8i;
typedef __attribute__((ext_vector_type(4)))  int v4i;

#define WMMA_F32_F16(a, b, c) \
  __builtin_amdgcn_wmma_f32_16x16x32_f16(false, (a), false, (b), (short)0, (c), false, false)

constexpr int Bsz = 2, S = 2048, D = 1024, Hh = 16, DH = 64;

__device__ inline unsigned lds_off(const void* p) {
  // addrspace(3) generic pointers carry the LDS offset in the low 32 bits
  return (unsigned)(unsigned long long)(uintptr_t)p;
}

// ---------------------------------------------------------------------------
// 2D f16 tile load Global->LDS via Tensor Data Mover.
// D# per CDNA5 ISA:
//  group0: [1:0]=count=1, [63:32]=lds_addr, [120:64]=global tile addr, [127:126]=type=2
//  group1: [17:16]=data_size(1=2B), [20]=pad_enable, [24:22]=pad_interval code,
//          [31:25]=pad_amount code, [79:48]=tensor_dim0, [111:80]=tensor_dim1,
//          [127:112]=tile_dim0, [143:128]=tile_dim1, [207:160]=tensor_dim0_stride
// padIntervalCode: DWORDs per row: 3 -> 16 DW (32 halves), 4 -> 32 DW (64 halves)
// padAmountCode:   3 -> 4 DW (8 halves of padding)
// ---------------------------------------------------------------------------
__device__ inline void tdm_load_2d_f16(const _Float16* gtile, unsigned ldsByteOff,
                                       unsigned tileRows, unsigned tileCols,
                                       unsigned tenRows, unsigned tenCols,
                                       unsigned rowStrideElems,
                                       unsigned padIntervalCode, unsigned padAmountCode) {
  unsigned long long ga = (unsigned long long)(uintptr_t)gtile;
  v4u g0;
  g0[0] = 1u;                                                  // count=1 (user, valid)
  g0[1] = ldsByteOff;                                          // lds_addr
  g0[2] = (unsigned)ga;                                        // global_addr[31:0]
  g0[3] = (unsigned)((ga >> 32) & 0x01ffffffu) | (2u << 30);   // addr[56:32] | type=2
  v8i g1;
  g1[0] = (int)((1u << 16) | (1u << 20) | (padIntervalCode << 22) | (padAmountCode << 25));
  g1[1] = (int)((tenCols & 0xffffu) << 16);                    // tensor_dim0[15:0]
  g1[2] = (int)(((tenCols >> 16) & 0xffffu) | ((tenRows & 0xffffu) << 16));
  g1[3] = (int)(((tenRows >> 16) & 0xffffu) | ((tileCols & 0xffffu) << 16));
  g1[4] = (int)(tileRows & 0xffffu);                           // tile_dim1 (tile_dim2=0)
  g1[5] = (int)rowStrideElems;                                 // tensor_dim0_stride[31:0]
  g1[6] = 0;
  g1[7] = 0;
  v4i z4 = {0, 0, 0, 0};
  v8i z8 = {0, 0, 0, 0, 0, 0, 0, 0};
  __builtin_amdgcn_tensor_load_to_lds(g0, g1, z4, z4, z8, 0);
}

// ---------------------------------------------------------------------------
// f32 -> f16 conversion
// ---------------------------------------------------------------------------
__global__ void cvt_f32_f16(const float* __restrict__ src, _Float16* __restrict__ dst, int n) {
  int i = blockIdx.x * blockDim.x + threadIdx.x;
  int stride = gridDim.x * blockDim.x;
  for (; i < n; i += stride) dst[i] = (_Float16)src[i];
}

// ---------------------------------------------------------------------------
// QKV projection: C[m,e] = sum_d X[m,d] * W[e,d]   (W stored (out,in))
// Block tile 128(M) x 64(N) x 32(K); TDM-staged LDS tiles; 8 waves, 32x32/wave.
// ---------------------------------------------------------------------------
__global__ __launch_bounds__(256) void qkv_gemm(
    const _Float16* __restrict__ Xh,
    const _Float16* __restrict__ Wqh,
    const _Float16* __restrict__ Wkh,
    const _Float16* __restrict__ Wvh,
    _Float16* __restrict__ Qh,
    _Float16* __restrict__ Kh,
    _Float16* __restrict__ Vh) {
  const _Float16* W = (blockIdx.z == 0) ? Wqh : (blockIdx.z == 1) ? Wkh : Wvh;
  _Float16* Out     = (blockIdx.z == 0) ? Qh  : (blockIdx.z == 1) ? Kh  : Vh;

  __shared__ _Float16 sA[128 * 40];  // 128 x 32, stride 40 (pad via TDM)
  __shared__ _Float16 sB[64 * 40];

  const int tid  = threadIdx.x;
  const int lane = tid & 31, wid = tid >> 5;
  const int wm = wid & 3, wn = wid >> 2;
  const int mbase = blockIdx.x * 128;
  const int nbase = blockIdx.y * 64;
  const unsigned sAoff = lds_off(sA), sBoff = lds_off(sB);

  v8f acc[2][2] = {};

  for (int k0 = 0; k0 < D; k0 += 32) {
    __syncthreads();                       // prior iter done reading sA/sB
    if (wid == 0) {
      tdm_load_2d_f16(&Xh[(size_t)mbase * D + k0], sAoff, 128, 32, Bsz * S, D, D, 3, 3);
      tdm_load_2d_f16(&W[(size_t)nbase * D + k0], sBoff, 64, 32, D, D, D, 3, 3);
      __builtin_amdgcn_s_wait_tensorcnt(0);
    }
    __syncthreads();

    const int g8  = (lane >> 4) * 8;
    const int g16 = (lane >> 4) * 16;
    const int rl  = lane & 15;

    v16h af[2], bf[2];
#pragma unroll
    for (int mi = 0; mi < 2; ++mi) {
      const _Float16* p = &sA[(wm * 32 + mi * 16 + rl) * 40];
#pragma unroll
      for (int i = 0; i < 16; ++i) af[mi][i] = p[g16 + i];
    }
#pragma unroll
    for (int ni = 0; ni < 2; ++ni) {
      const _Float16* p = &sB[(wn * 32 + ni * 16 + rl) * 40];
#pragma unroll
      for (int i = 0; i < 16; ++i) bf[ni][i] = p[g16 + i];
    }
#pragma unroll
    for (int mi = 0; mi < 2; ++mi)
#pragma unroll
      for (int ni = 0; ni < 2; ++ni)
        acc[mi][ni] = WMMA_F32_F16(af[mi], bf[ni], acc[mi][ni]);
  }

  const int rowoff = (lane >> 4) * 8;
  const int coloff = lane & 15;
#pragma unroll
  for (int mi = 0; mi < 2; ++mi)
#pragma unroll
    for (int ni = 0; ni < 2; ++ni)
#pragma unroll
      for (int r = 0; r < 8; ++r) {
        int gm = mbase + wm * 32 + mi * 16 + r + rowoff;
        int gn = nbase + wn * 32 + ni * 16 + coloff;
        int b = gm >> 11, s = gm & 2047, h = gn >> 6, dh = gn & 63;
        Out[((size_t)((b << 4) + h) * S + s) * DH + dh] = (_Float16)acc[mi][ni][r];
      }
}

// ---------------------------------------------------------------------------
// Flash attention: one (b,h) per blockIdx.y, 128 queries per blockIdx.x.
// Q/K tiles via TDM; V staged TRANSPOSED ([dh][key]) so P@V B-fragments are
// contiguous (2x ds_load_b128 instead of 16 strided u16 loads).
// ---------------------------------------------------------------------------
__global__ __launch_bounds__(256) void flash_attn(
    const _Float16* __restrict__ Qh,
    const _Float16* __restrict__ Kh,
    const _Float16* __restrict__ Vh,
    _Float16* __restrict__ CtxH) {
  __shared__ _Float16 sQ[128 * 72];
  __shared__ _Float16 sK[64 * 72];
  __shared__ _Float16 sVt[64 * 72];      // transposed: [dh][key], stride 72
  __shared__ _Float16 sP[8 * 16 * 72];

  const int tid  = threadIdx.x;
  const int lane = tid & 31, w = tid >> 5;
  const int bh     = blockIdx.y;         // b*16 + h
  const int qblock = blockIdx.x;
  const int qbase  = qblock * 128;

  const _Float16* Qp = Qh + (size_t)bh * S * DH;
  const _Float16* Kp = Kh + (size_t)bh * S * DH;
  const _Float16* Vp = Vh + (size_t)bh * S * DH;
  const unsigned sQoff = lds_off(sQ), sKoff = lds_off(sK);

  if (w == 0) {
    tdm_load_2d_f16(Qp + (size_t)qbase * DH, sQoff, 128, 64, S, DH, DH, 4, 3);
    __builtin_amdgcn_s_wait_tensorcnt(0);
  }
  __syncthreads();

  const int g8  = (lane >> 4) * 8;
  const int g16 = (lane >> 4) * 16;
  const int rl  = lane & 15;
  const int row8 = (lane >> 4) * 8;      // C-layout row offset

  v16h qf[2];
#pragma unroll
  for (int c = 0; c < 2; ++c) {
    const _Float16* p = &sQ[(w * 16 + rl) * 72 + c * 32];
#pragma unroll
    for (int i = 0; i < 16; ++i) qf[c][i] = p[g16 + i];
  }

  v8f acc[4] = {};
  float m_run[8], l_run[8];
#pragma unroll
  for (int r = 0; r < 8; ++r) { m_run[r] = -1e30f; l_run[r] = 0.f; }

  const int nkb = 2 * qblock + 2;        // causal: keys <= qbase+127
  for (int kb = 0; kb < nkb; ++kb) {
    const int kvbase = kb * 64;
    __syncthreads();                     // prior iter done reading sK/sVt
    if (w == 0)                          // K tile via TDM (overlaps V staging)
      tdm_load_2d_f16(Kp + (size_t)kvbase * DH, sKoff, 64, 64, S, DH, DH, 4, 3);
    // V tile staged transposed by all threads
#pragma unroll
    for (int it = 0; it < 2; ++it) {
      int idx = tid + it * 256;          // 0..511
      int r = idx >> 3, c = (idx & 7) * 8;
      v8h vv = *(const v8h*)&Vp[(size_t)(kvbase + r) * DH + c];
#pragma unroll
      for (int i = 0; i < 8; ++i) sVt[(c + i) * 72 + r] = vv[i];
    }
    if (w == 0) __builtin_amdgcn_s_wait_tensorcnt(0);
    __syncthreads();

    // S = Q @ K^T  (4 key tiles x 2 K-chunks)
    v8f st[4];
#pragma unroll
    for (int t = 0; t < 4; ++t) {
      v8f z = {};
#pragma unroll
      for (int c = 0; c < 2; ++c) {
        v16h kf;
        const _Float16* p = &sK[(t * 16 + rl) * 72 + c * 32];
#pragma unroll
        for (int i = 0; i < 16; ++i) kf[i] = p[g16 + i];
        z = WMMA_F32_F16(qf[c], kf, z);
      }
      st[t] = z;
    }

    // causal mask + 1/sqrt(DH)
#pragma unroll
    for (int t = 0; t < 4; ++t)
#pragma unroll
      for (int r = 0; r < 8; ++r) {
        int q = qbase + w * 16 + r + row8;
        int k = kvbase + t * 16 + rl;
        float v = st[t][r] * 0.125f;
        st[t][r] = (k > q) ? -1e30f : v;
      }

    // online softmax (per-row stats; xor masks < 16 stay within lane-half)
#pragma unroll
    for (int r = 0; r < 8; ++r) {
      float mx = st[0][r];
#pragma unroll
      for (int t = 1; t < 4; ++t) mx = fmaxf(mx, st[t][r]);
#pragma unroll
      for (int d = 1; d < 16; d <<= 1) mx = fmaxf(mx, __shfl_xor(mx, d, 32));
      float m_new = fmaxf(m_run[r], mx);
      float corr = __expf(m_run[r] - m_new);
      m_run[r] = m_new;
      l_run[r] *= corr;
#pragma unroll
      for (int t = 0; t < 4; ++t) acc[t][r] *= corr;
      float rs = 0.f;
#pragma unroll
      for (int t = 0; t < 4; ++t) {
        float p = __expf(st[t][r] - m_new);
        st[t][r] = p;
        rs += p;
      }
#pragma unroll
      for (int d = 1; d < 16; d <<= 1) rs += __shfl_xor(rs, d, 32);
      l_run[r] += rs;
    }

    // transpose P through per-wave LDS into A-fragment layout
    _Float16* pw = &sP[w * 16 * 72];
#pragma unroll
    for (int t = 0; t < 4; ++t)
#pragma unroll
      for (int r = 0; r < 8; ++r)
        pw[(r + row8) * 72 + t * 16 + rl] = (_Float16)st[t][r];
    // same-wave DS ops are in-order; no barrier needed for pw reuse

    v16h pf[2];
#pragma unroll
    for (int c = 0; c < 2; ++c) {
      const _Float16* p = &pw[rl * 72 + c * 32];
#pragma unroll
      for (int i = 0; i < 16; ++i) pf[c][i] = p[g16 + i];
    }
    // ctx += P @ V : B fragment contiguous from transposed sVt
#pragma unroll
    for (int t = 0; t < 4; ++t) {
#pragma unroll
      for (int c = 0; c < 2; ++c) {
        v16h vf;
        const _Float16* pv = &sVt[(t * 16 + rl) * 72 + c * 32 + g16];
#pragma unroll
        for (int i = 0; i < 16; ++i) vf[i] = pv[i];
        acc[t] = WMMA_F32_F16(pf[c], vf, acc[t]);
      }
    }
  }

  // normalize + store ctx as f16 row-major [4096][1024], col = h*64+dh
  const int b = bh >> 4, h = bh & 15;
#pragma unroll
  for (int t = 0; t < 4; ++t)
#pragma unroll
    for (int r = 0; r < 8; ++r) {
      int s = qbase + w * 16 + r + row8;
      int col = h * DH + t * 16 + rl;
      CtxH[(size_t)(b * S + s) * D + col] = (_Float16)(acc[t][r] / l_run[r]);
    }
}

// ---------------------------------------------------------------------------
// Output projection: out[m,e] = sum_d ctx[m,d] * wo[e,d] + bo[e]  (f32 out)
// ---------------------------------------------------------------------------
__global__ __launch_bounds__(256) void out_gemm(
    const _Float16* __restrict__ Ah,
    const _Float16* __restrict__ Wh,
    const float* __restrict__ bias,
    float* __restrict__ Out) {
  __shared__ _Float16 sA[128 * 40];
  __shared__ _Float16 sB[64 * 40];

  const int tid  = threadIdx.x;
  const int lane = tid & 31, wid = tid >> 5;
  const int wm = wid & 3, wn = wid >> 2;
  const int mbase = blockIdx.x * 128;
  const int nbase = blockIdx.y * 64;
  const unsigned sAoff = lds_off(sA), sBoff = lds_off(sB);

  v8f acc[2][2] = {};

  for (int k0 = 0; k0 < D; k0 += 32) {
    __syncthreads();
    if (wid == 0) {
      tdm_load_2d_f16(&Ah[(size_t)mbase * D + k0], sAoff, 128, 32, Bsz * S, D, D, 3, 3);
      tdm_load_2d_f16(&Wh[(size_t)nbase * D + k0], sBoff, 64, 32, D, D, D, 3, 3);
      __builtin_amdgcn_s_wait_tensorcnt(0);
    }
    __syncthreads();

    const int g8  = (lane >> 4) * 8;
    const int g16 = (lane >> 4) * 16;
    const int rl  = lane & 15;

    v16h af[2], bf[2];
#pragma unroll
    for (int mi = 0; mi < 2; ++mi) {
      const _Float16* p = &sA[(wm * 32 + mi * 16 + rl) * 40];
#pragma unroll
      for (int i = 0; i < 16; ++i) af[mi][i] = p[g16 + i];
    }
#pragma unroll
    for (int ni = 0; ni < 2; ++ni) {
      const _Float16* p = &sB[(wn * 32 + ni * 16 + rl) * 40];
#pragma unroll
      for (int i = 0; i < 16; ++i) bf[ni][i] = p[g16 + i];
    }
#pragma unroll
    for (int mi = 0; mi < 2; ++mi)
#pragma unroll
      for (int ni = 0; ni < 2; ++ni)
        acc[mi][ni] = WMMA_F32_F16(af[mi], bf[ni], acc[mi][ni]);
  }

  const int rowoff = (lane >> 4) * 8;
  const int coloff = lane & 15;
#pragma unroll
  for (int mi = 0; mi < 2; ++mi)
#pragma unroll
    for (int ni = 0; ni < 2; ++ni)
#pragma unroll
      for (int r = 0; r < 8; ++r) {
        int gm = mbase + wm * 32 + mi * 16 + r + rowoff;
        int gn = nbase + wn * 32 + ni * 16 + coloff;
        Out[(size_t)gm * D + gn] = acc[mi][ni][r] + bias[gn];
      }
}

// ---------------------------------------------------------------------------
extern "C" void kernel_launch(void* const* d_in, const int* in_sizes, int n_in,
                              void* d_out, int out_size, void* d_ws, size_t ws_size,
                              hipStream_t stream) {
  const float* X  = (const float*)d_in[0];
  const float* wq = (const float*)d_in[1];
  const float* wk = (const float*)d_in[2];
  const float* wv = (const float*)d_in[3];
  const float* wo = (const float*)d_in[4];
  const float* bo = (const float*)d_in[5];
  float* out = (float*)d_out;

  const size_t NX = (size_t)Bsz * S * D;       // 4,194,304
  const size_t NW = (size_t)D * D;             // 1,048,576
  const size_t NQ = (size_t)Bsz * Hh * S * DH; // 4,194,304

  _Float16* ws  = (_Float16*)d_ws;
  _Float16* Xh  = ws;
  _Float16* Wqh = Xh + NX;
  _Float16* Wkh = Wqh + NW;
  _Float16* Wvh = Wkh + NW;
  _Float16* Woh = Wvh + NW;
  _Float16* Qh  = Woh + NW;
  _Float16* Kh  = Qh + NQ;
  _Float16* Vh  = Kh + NQ;
  _Float16* Ctx = Vh + NQ;

  cvt_f32_f16<<<1024, 256, 0, stream>>>(X,  Xh,  (int)NX);
  cvt_f32_f16<<<512,  256, 0, stream>>>(wq, Wqh, (int)NW);
  cvt_f32_f16<<<512,  256, 0, stream>>>(wk, Wkh, (int)NW);
  cvt_f32_f16<<<512,  256, 0, stream>>>(wv, Wvh, (int)NW);
  cvt_f32_f16<<<512,  256, 0, stream>>>(wo, Woh, (int)NW);

  dim3 g1(32, 16, 3);                 // 4096/128, 1024/64, {q,k,v}
  qkv_gemm<<<g1, 256, 0, stream>>>(Xh, Wqh, Wkh, Wvh, Qh, Kh, Vh);

  dim3 g2(16, 32);                    // 2048/128 query blocks, B*H
  flash_attn<<<g2, 256, 0, stream>>>(Qh, Kh, Vh, Ctx);

  dim3 g3(32, 16);
  out_gemm<<<g3, 256, 0, stream>>>(Ctx, Woh, bo, out);
}
